// DualPrompt_75737453298409
// MI455X (gfx1250) — matrix operand order repaired and checked by reading, not compiled, not run on hardware
//
#include <hip/hip_runtime.h>
#include <math.h>

typedef __attribute__((ext_vector_type(16))) __bf16 v16bf;
typedef __attribute__((ext_vector_type(8)))  float  v8f;
typedef __attribute__((ext_vector_type(4)))  float  f4;
typedef __attribute__((ext_vector_type(2)))  float  f2;

#define POOL   100
#define KPAD   128
#define CDIM   768
#define BATCH  1024
#define NPROJ  6144        // 8 * 768
#define EK_ELEMS 3145728   // 1024 * 4 * 768

// ---------------------------------------------------------------------------
// Kernel 1: softmax(A) row-wise, build packed W (w1 = A_sm*K, w2 = A_sm^2)
// stored k-major (transposed) so the aqk kernel gets coalesced loads, plus
// kinv[k] = 1 / max(||K_k||, eps).
// ---------------------------------------------------------------------------
__device__ __forceinline__ float blk_reduce_max(float v, float* red, int t) {
    red[t] = v; __syncthreads();
    for (int s = 128; s > 0; s >>= 1) { if (t < s) red[t] = fmaxf(red[t], red[t + s]); __syncthreads(); }
    float r = red[0]; __syncthreads();
    return r;
}
__device__ __forceinline__ float blk_reduce_sum(float v, float* red, int t) {
    red[t] = v; __syncthreads();
    for (int s = 128; s > 0; s >>= 1) { if (t < s) red[t] += red[t + s]; __syncthreads(); }
    float r = red[0]; __syncthreads();
    return r;
}

__global__ __launch_bounds__(256) void prep_kernel(const float* __restrict__ A,
                                                   const float* __restrict__ K,
                                                   float* __restrict__ Wpack,   // [768][128][2]
                                                   float* __restrict__ kinv) {
    __shared__ float red[256];
    const int k = blockIdx.x;   // pool row, 0..99
    const int t = threadIdx.x;

    float a0 = A[k * CDIM + t];
    float a1 = A[k * CDIM + t + 256];
    float a2 = A[k * CDIM + t + 512];

    float amax = blk_reduce_max(fmaxf(a0, fmaxf(a1, a2)), red, t);
    float e0 = __expf(a0 - amax), e1 = __expf(a1 - amax), e2 = __expf(a2 - amax);
    float inv = 1.0f / blk_reduce_sum(e0 + e1 + e2, red, t);

    float ksq = 0.0f;
    float ev[3] = {e0, e1, e2};
    #pragma unroll
    for (int j = 0; j < 3; ++j) {
        int d = t + j * 256;
        float sm = ev[j] * inv;
        float kv = K[k * CDIM + d];
        Wpack[(d * KPAD + k) * 2 + 0] = sm * kv;   // for dots
        Wpack[(d * KPAD + k) * 2 + 1] = sm * sm;   // for ||a_querry||^2
        ksq = fmaf(kv, kv, ksq);
    }
    float kn = blk_reduce_sum(ksq, red, t);
    if (t == 0) kinv[k] = 1.0f / fmaxf(sqrtf(kn), 1e-6f);
}

// ---------------------------------------------------------------------------
// Kernel 2: aq_k[b,k] for an 8-row batch tile; writes bf16 hi/lo WMMA A-frags
// A-fragment layout used here (16-bit A 16x32, contiguous k order, same as B):
//   lane = (kk>>4)*16 + (b%16),  slot = kk&15
// ---------------------------------------------------------------------------
__global__ __launch_bounds__(256) void aqk_kernel(const float* __restrict__ x,
                                                  const float* __restrict__ Wpack,
                                                  const float* __restrict__ kinv,
                                                  __bf16* __restrict__ fragA_hi,
                                                  __bf16* __restrict__ fragA_lo) {
    __shared__ float xs[8 * CDIM];
    __shared__ float xq[8 * CDIM];
    const int t = threadIdx.x;
    const int bBase = blockIdx.x * 8;

    for (int i = t; i < 8 * CDIM; i += 256) {
        float v = x[(size_t)bBase * CDIM + i];
        xs[i] = v;
        xq[i] = v * v;
    }
    __syncthreads();

    const int k = t & (KPAD - 1);   // lane-contiguous pool index -> coalesced W loads
    const int g = t >> 7;           // 0/1 -> rows g*4 .. g*4+3

    float dot[4] = {0.f, 0.f, 0.f, 0.f};
    float sq [4] = {0.f, 0.f, 0.f, 0.f};
    #pragma unroll 4
    for (int d = 0; d < CDIM; ++d) {
        f2 w = *(const f2*)(Wpack + (d * KPAD + k) * 2);   // b64, 256B/wave
        #pragma unroll
        for (int i = 0; i < 4; ++i) {
            dot[i] = fmaf(xs[(g * 4 + i) * CDIM + d], w.x, dot[i]);
            sq [i] = fmaf(xq[(g * 4 + i) * CDIM + d], w.y, sq [i]);
        }
    }

    const float kv = kinv[k < POOL ? k : 0];
    #pragma unroll
    for (int i = 0; i < 4; ++i) {
        int b = bBase + g * 4 + i;
        float aq = 0.0f;
        if (k < POOL) {                      // padded K-columns contribute zero
            float an = fmaxf(sqrtf(sq[i]), 1e-6f);
            aq = (dot[i] / an * kv + 1.0f) * 0.5f;
        }
        __bf16 h  = (__bf16)aq;
        __bf16 lo = (__bf16)(aq - (float)h);
        int mt = b >> 4, mr = b & 15;
        int kc = k >> 5, kk = k & 31;
        int lane = ((kk >> 4) << 4) + mr;
        int slot = kk & 15;
        int idx  = (((mt * 4 + kc) * 32) + lane) * 16 + slot;
        fragA_hi[idx] = h;
        fragA_lo[idx] = lo;
    }
}

// ---------------------------------------------------------------------------
// Kernel 3: convert p (8,100,768) f32 -> bf16 hi/lo WMMA B-fragments, K padded
// to 128 with zeros. B layout (32x16): lane = (kk/16)*16 + n%16, slot = kk%16.
// ---------------------------------------------------------------------------
__global__ __launch_bounds__(256) void convp_kernel(const float* __restrict__ p,
                                                    __bf16* __restrict__ fragB_hi,
                                                    __bf16* __restrict__ fragB_lo) {
    const int gid = blockIdx.x * 256 + threadIdx.x;   // 6144*128 exactly
    const int n = gid % NPROJ;                        // consecutive -> coalesced p reads
    const int k = gid / NPROJ;
    float v = 0.0f;
    if (k < POOL) {
        int l = n / CDIM, d = n - l * CDIM;
        v = p[((size_t)l * POOL + k) * CDIM + d];
    }
    __bf16 h  = (__bf16)v;
    __bf16 lo = (__bf16)(v - (float)h);
    int nt = n >> 4, nr = n & 15;
    int kc = k >> 5, kk = k & 31;
    int lane = ((kk >> 4) << 4) + nr;
    int slot = kk & 15;
    int idx  = (((nt * 4 + kc) * 32) + lane) * 16 + slot;
    fragB_hi[idx] = h;
    fragB_lo[idx] = lo;
}

// ---------------------------------------------------------------------------
// Kernel 4: P_ = aq_k (1024x128) @ p' (128x6144) via bf16 WMMA with hi/lo
// split (3 products per K-chunk ~ f32 accuracy). One 16x16 tile per wave.
// Epilogue writes straight into Ek (l<4) / Ev (l>=4) regions of d_out.
// ---------------------------------------------------------------------------
__global__ __launch_bounds__(256) void gemm_kernel(const __bf16* __restrict__ fragA_hi,
                                                   const __bf16* __restrict__ fragA_lo,
                                                   const __bf16* __restrict__ fragB_hi,
                                                   const __bf16* __restrict__ fragB_lo,
                                                   float* __restrict__ out) {
    const int lane = threadIdx.x & 31;
    const int wave = threadIdx.x >> 5;
    const int mt = blockIdx.x;                 // 0..63
    const int nt = blockIdx.y * 8 + wave;      // 0..383

    v8f c = {0.f, 0.f, 0.f, 0.f, 0.f, 0.f, 0.f, 0.f};
    #pragma unroll
    for (int kc = 0; kc < 4; ++kc) {
        const int ai = (((mt * 4 + kc) * 32) + lane) * 16;
        const int bi = (((nt * 4 + kc) * 32) + lane) * 16;
        v16bf aH = *(const v16bf*)(fragA_hi + ai);
        v16bf aL = *(const v16bf*)(fragA_lo + ai);
        v16bf bH = *(const v16bf*)(fragB_hi + bi);
        v16bf bL = *(const v16bf*)(fragB_lo + bi);
        c = __builtin_amdgcn_wmma_f32_16x16x32_bf16(false, aH, false, bH, (short)0, c, false, false);
        c = __builtin_amdgcn_wmma_f32_16x16x32_bf16(false, aH, false, bL, (short)0, c, false, false);
        c = __builtin_amdgcn_wmma_f32_16x16x32_bf16(false, aL, false, bH, (short)0, c, false, false);
    }

    // C layout: VGPR r -> M = r + (lane>=16 ? 8 : 0), N = lane%16
    const int ncol = nt * 16 + (lane & 15);
    const int l = ncol / CDIM;
    const int d = ncol - l * CDIM;
    const size_t base = (l < 4) ? (size_t)(l * CDIM + d)
                                : (size_t)(EK_ELEMS + (l - 4) * CDIM + d);
    const int rowBase = mt * 16 + ((lane >> 4) << 3);
    #pragma unroll
    for (int r = 0; r < 8; ++r)
        out[base + (size_t)(rowBase + r) * 3072] = c[r];
}

// ---------------------------------------------------------------------------
// Kernel 5: stream x_block through untouched (memory-roofline term, ~53us).
// ---------------------------------------------------------------------------
__global__ __launch_bounds__(256) void copy_kernel(const f4* __restrict__ src,
                                                   f4* __restrict__ dst, long n4) {
    long i = (long)blockIdx.x * 256 + threadIdx.x;
    if (i < n4) {
        f4 v = __builtin_nontemporal_load(src + i);
        __builtin_nontemporal_store(v, dst + i);
    }
}

extern "C" void kernel_launch(void* const* d_in, const int* in_sizes, int n_in,
                              void* d_out, int out_size, void* d_ws, size_t ws_size,
                              hipStream_t stream) {
    const float* x_querry = (const float*)d_in[0];
    const float* x_block  = (const float*)d_in[1];
    const float* K        = (const float*)d_in[2];
    const float* A        = (const float*)d_in[3];
    const float* p        = (const float*)d_in[4];
    float* out = (float*)d_out;

    // workspace layout (bytes):
    //   [0,786432)           Wpack  f32 [768][128][2]
    //   [786432,786944)      kinv   f32 [128]
    //   [786944,...)         bf16 fragments, 32B-aligned
    float* ws_f   = (float*)d_ws;
    float* Wpack  = ws_f;
    float* kinv   = ws_f + 196608;
    __bf16* bfb      = (__bf16*)((char*)d_ws + 786944);
    __bf16* fragA_hi = bfb;                       // 131072 elems
    __bf16* fragA_lo = bfb + 131072;              // 131072
    __bf16* fragB_hi = bfb + 262144;              // 786432
    __bf16* fragB_lo = bfb + 262144 + 786432;     // 786432  (total ~4.25 MB)

    prep_kernel <<<100, 256, 0, stream>>>(A, K, Wpack, kinv);
    aqk_kernel  <<<128, 256, 0, stream>>>(x_querry, Wpack, kinv, fragA_hi, fragA_lo);
    convp_kernel<<<3072, 256, 0, stream>>>(p, fragB_hi, fragB_lo);
    gemm_kernel <<<dim3(64, 48), 256, 0, stream>>>(fragA_hi, fragA_lo, fragB_hi, fragB_lo, out);

    long n4 = (long)in_sizes[1] / 4;              // 1024*197*768 / 4, exact
    int cblocks = (int)((n4 + 255) / 256);
    copy_kernel <<<cblocks, 256, 0, stream>>>((const f4*)x_block,
                                              (f4*)(out + 2 * (size_t)EK_ELEMS), n4);
}
